// MatchNet_3822520893581
// MI455X (gfx1250) — matrix operand order repaired and checked by reading, not compiled, not run on hardware
//
#include <hip/hip_runtime.h>
#include <math.h>

#define BB 16
#define LP 1024
#define LQ 1024
#define HH 1024

typedef __attribute__((ext_vector_type(16))) __bf16 v16bf;
typedef __attribute__((ext_vector_type(8)))  __bf16 v8bf;
typedef __attribute__((ext_vector_type(8)))  float  v8f;

__device__ __forceinline__ v8f wmma_bf16(v16bf a, v16bf b, v8f c) {
  return __builtin_amdgcn_wmma_f32_16x16x32_bf16(false, a, false, b,
                                                 (short)0, c, false, false);
}

// ---------------------------------------------------------------------------
// Kernel 0a: W[h][k] (fp32) -> Wt[k][h] (bf16): keys-GEMM B-frags become
// contiguous 32B loads per lane.
__global__ void __launch_bounds__(256) conv_w(const float* __restrict__ W,
                                              __bf16* __restrict__ Wt) {
  int idx = blockIdx.x * 256 + threadIdx.x;     // HH*HH elements
  int k = idx >> 10;
  int h = idx & (HH - 1);
  Wt[(size_t)k * HH + h] = (__bf16)W[(size_t)h * HH + k];
}

// Kernel 0b: q[b][qi][h] (fp32) -> qT[b][h][qi] (bf16): values-GEMM B-frags
// become contiguous 32B loads per lane.
__global__ void __launch_bounds__(256) conv_q(const float* __restrict__ q,
                                              __bf16* __restrict__ qT) {
  int idx = blockIdx.x * 256 + threadIdx.x;     // BB*HH*LQ elements
  int qi = idx & (LQ - 1);
  int t  = idx >> 10;
  int h  = t & (HH - 1);
  int b  = t >> 10;
  qT[(size_t)idx] = (__bf16)q[((size_t)(b * LQ + qi)) * HH + h];
}

// ---------------------------------------------------------------------------
// Kernel 1: keys[b][qi][k] = bf16( q[b][qi][:] @ W[:,k] + bias[k] )
// Block = 4 waves sharing one 16-row q-tile staged once in LDS (bf16);
// each wave computes a 16x64 output chunk (4 accumulators, 32 K-steps).
__global__ void __launch_bounds__(128) keys_kernel(const float* __restrict__ q,
                                                   const __bf16* __restrict__ Wt,
                                                   const float* __restrict__ bias,
                                                   __bf16* __restrict__ keys) {
  __shared__ __align__(16) __bf16 qb[16][HH];   // 32 KB

  const int tid  = threadIdx.x;
  const int lane = tid & 31;
  const int w    = tid >> 5;                    // wave 0..3
  const int row0 = blockIdx.y * 16;
  const int b    = blockIdx.z;
  const int col0 = blockIdx.x * 256 + w * 64;

  // stage q row-tile as bf16 (coalesced, shared by all 4 waves)
  for (int idx = tid; idx < 16 * HH; idx += 128) {
    int r = idx >> 10, c = idx & (HH - 1);
    qb[r][c] = (__bf16)q[(size_t)(b * LQ + row0 + r) * HH + c];
  }
  __syncthreads();

  const int M     = lane & 15;
  const int hi    = lane >> 4;
  const int koffA = hi * 16;    // A layout: K {0..15} / {16..31}
  const int koffB = hi * 16;    // B layout: K {0..15} / {16..31}
  const int n     = lane & 15;

  v8f acc[4];
  #pragma unroll
  for (int t = 0; t < 4; ++t) acc[t] = (v8f){};

  #pragma unroll 4
  for (int kb = 0; kb < HH; kb += 32) {
    v8bf alo = *(const v8bf*)(&qb[M][kb + koffA]);
    v8bf ahi = *(const v8bf*)(&qb[M][kb + 8 + koffA]);
    v16bf a;
    #pragma unroll
    for (int e = 0; e < 8; ++e) { a[e] = alo[e]; a[e + 8] = ahi[e]; }
    // WGP-scope prefetch (locality 3) of the streaming B operand
    __builtin_prefetch((const void*)(Wt + (size_t)(col0 + n) * HH + kb + 128), 0, 3);
    #pragma unroll
    for (int t = 0; t < 4; ++t) {
      const int col = col0 + t * 16 + n;
      v16bf bv = *(const v16bf*)(Wt + (size_t)col * HH + kb + koffB);
      acc[t] = wmma_bf16(a, bv, acc[t]);
    }
  }

  #pragma unroll
  for (int t = 0; t < 4; ++t) {
    const int col = col0 + t * 16 + n;
    const float bk = bias[col];
    #pragma unroll
    for (int r = 0; r < 8; ++r) {
      const int row = row0 + r + hi * 8;          // C layout rows r / r+8
      keys[(size_t)(b * LQ + row) * HH + col] = (__bf16)(acc[t][r] + bk);
    }
  }
}

// ---------------------------------------------------------------------------
// Score tile: S(16x16) = p_tile(16xH) @ keys_row_tile(16xH)^T.
__device__ __forceinline__ v8f score_tile(const __bf16* pb,         // LDS [16][HH]
                                          const __bf16* krow,       // keys row base (+koffB)
                                          int M, int koffA) {
  v8f s = (v8f){};
  #pragma unroll 4
  for (int kb = 0; kb < HH; kb += 32) {
    __builtin_prefetch((const void*)(krow + kb + 128), 0, 3);
    v8bf alo = *(const v8bf*)(pb + (size_t)M * HH + kb + koffA);
    v8bf ahi = *(const v8bf*)(pb + (size_t)M * HH + kb + 8 + koffA);
    v16bf a;
    #pragma unroll
    for (int e = 0; e < 8; ++e) { a[e] = alo[e]; a[e + 8] = ahi[e]; }
    v16bf bv = *(const v16bf*)(krow + kb);
    s = wmma_bf16(a, bv, s);
  }
  return s;
}

// Kernel 2: fused attention. Block = 4 waves cooperating on one 16-row p-tile:
//  pass1: each wave covers 16 of the 64 key tiles, online row max/sum
//         (cross-wave merge through a 512B LDS scratch aliased into Pm)
//  pass2: recompute own score tiles, write P = exp(s-m)/l (bf16) into LDS
//  pass3: waves split the 16 output 64-column chunks of relu(P @ q)
__global__ void __launch_bounds__(128) attn_kernel(const float* __restrict__ p,
                                                   const __bf16* __restrict__ qT,
                                                   const __bf16* __restrict__ keys,
                                                   float* __restrict__ out) {
  __shared__ __align__(16) __bf16 pb[16][HH];   // 32 KB: bf16 p-tile
  __shared__ __align__(16) __bf16 Pm[16][LQ];   // 32 KB: normalized probs

  const int tid   = threadIdx.x;
  const int lane  = tid & 31;
  const int w     = tid >> 5;
  const int prow0 = blockIdx.x * 16;
  const int b     = blockIdx.y;
  const int M     = lane & 15;
  const int hi    = lane >> 4;
  const int koffA = hi * 16;
  const int koffB = hi * 16;
  const int n     = lane & 15;

  // ---- pass 0: stage p-tile into LDS as bf16 (coalesced, all threads) ----
  for (int idx = tid; idx < 16 * HH; idx += 128) {
    int r = idx >> 10, c = idx & (HH - 1);
    pb[r][c] = (__bf16)p[(size_t)(b * LP + prow0 + r) * HH + c];
  }
  __syncthreads();

  const __bf16* kbase = keys + (size_t)b * LQ * HH;

  // ---- pass 1: per-wave online softmax stats over 16 of 64 key tiles ----
  float m[8], l[8];
  #pragma unroll
  for (int r = 0; r < 8; ++r) { m[r] = -INFINITY; l[r] = 0.f; }

  for (int j = w * 16; j < LQ; j += 64) {
    const __bf16* krow = kbase + (size_t)(j + n) * HH + koffB;
    v8f s = score_tile(&pb[0][0], krow, M, koffA);
    #pragma unroll
    for (int r = 0; r < 8; ++r) {
      float v  = s[r];
      float rm = v;
      #pragma unroll
      for (int mk = 1; mk <= 8; mk <<= 1)        // reduce across 16-lane group
        rm = fmaxf(rm, __shfl_xor(rm, mk, 32));
      float mnew = fmaxf(m[r], rm);
      float e    = __expf(v - mnew);
      #pragma unroll
      for (int mk = 1; mk <= 8; mk <<= 1)
        e += __shfl_xor(e, mk, 32);
      l[r] = l[r] * __expf(m[r] - mnew) + e;
      m[r] = mnew;
    }
  }

  // ---- cross-wave stats merge via 512B scratch aliased into Pm ----
  // Pm is not live yet; stats are consumed before pass2 overwrites it.
  float* smf = (float*)&Pm[0][0];   // [0..63]=m per (wave,row), [64..127]=l
  if (n == 0) {                      // lanes 0 and 16 (rows 0-7 / 8-15)
    #pragma unroll
    for (int r = 0; r < 8; ++r) {
      smf[w * 16 + r + hi * 8]      = m[r];
      smf[64 + w * 16 + r + hi * 8] = l[r];
    }
  }
  __syncthreads();

  float linv[8];
  #pragma unroll
  for (int r = 0; r < 8; ++r) {
    const int row = r + hi * 8;
    float m0 = smf[row],      m1 = smf[16 + row];
    float m2 = smf[32 + row], m3 = smf[48 + row];
    float mg = fmaxf(fmaxf(m0, m1), fmaxf(m2, m3));
    float lg = smf[64 + row]      * __expf(m0 - mg)
             + smf[64 + 16 + row] * __expf(m1 - mg)
             + smf[64 + 32 + row] * __expf(m2 - mg)
             + smf[64 + 48 + row] * __expf(m3 - mg);
    m[r]    = mg;
    linv[r] = 1.f / lg;
  }
  __syncthreads();   // everyone has read stats; Pm may now be overwritten

  // ---- pass 2: recompute own score tiles, write normalized probs ----
  for (int j = w * 16; j < LQ; j += 64) {
    const __bf16* krow = kbase + (size_t)(j + n) * HH + koffB;
    v8f s = score_tile(&pb[0][0], krow, M, koffA);
    #pragma unroll
    for (int r = 0; r < 8; ++r)
      Pm[r + hi * 8][j + n] = (__bf16)(__expf(s[r] - m[r]) * linv[r]);
  }
  __syncthreads();

  // ---- pass 3: out = relu(P @ q); waves split the 64-column chunks ----
  const __bf16* qTb = qT + (size_t)b * HH * LQ;
  for (int nc = w * 64; nc < HH; nc += 256) {
    v8f acc[4];
    #pragma unroll
    for (int t = 0; t < 4; ++t) acc[t] = (v8f){};

    #pragma unroll 2
    for (int kb = 0; kb < LQ; kb += 32) {
      v8bf alo = *(const v8bf*)(&Pm[M][kb + koffA]);
      v8bf ahi = *(const v8bf*)(&Pm[M][kb + 8 + koffA]);
      v16bf a;
      #pragma unroll
      for (int e = 0; e < 8; ++e) { a[e] = alo[e]; a[e + 8] = ahi[e]; }
      __builtin_prefetch((const void*)(qTb + (size_t)(nc + n) * LQ + kb + 128), 0, 3);
      #pragma unroll
      for (int t = 0; t < 4; ++t) {
        const int col = nc + t * 16 + n;
        v16bf bv = *(const v16bf*)(qTb + (size_t)col * LQ + kb + koffB);
        acc[t] = wmma_bf16(a, bv, acc[t]);
      }
    }
    #pragma unroll
    for (int t = 0; t < 4; ++t) {
      const int col = nc + t * 16 + n;
      #pragma unroll
      for (int r = 0; r < 8; ++r) {
        const int row = prow0 + r + hi * 8;
        float v = acc[t][r];
        out[(size_t)(b * LP + row) * HH + col] = v > 0.f ? v : 0.f;
      }
    }
  }
}

// ---------------------------------------------------------------------------
extern "C" void kernel_launch(void* const* d_in, const int* in_sizes, int n_in,
                              void* d_out, int out_size, void* d_ws, size_t ws_size,
                              hipStream_t stream) {
  const float* p    = (const float*)d_in[0];   // [B, LP, H]
  const float* q    = (const float*)d_in[1];   // [B, LQ, H]
  const float* W    = (const float*)d_in[2];   // [H, H]
  const float* bias = (const float*)d_in[3];   // [H]
  float* out = (float*)d_out;

  // Workspace (bf16): keys [B*LQ*H] (32MB) | Wt [H*H] (2MB) | qT [B*H*LQ] (32MB)
  __bf16* keys = (__bf16*)d_ws;
  __bf16* Wt   = keys + (size_t)BB * LQ * HH;
  __bf16* qT   = Wt + (size_t)HH * HH;

  conv_w<<<(HH * HH) / 256, 256, 0, stream>>>(W, Wt);
  conv_q<<<(BB * HH * LQ) / 256, 256, 0, stream>>>(q, qT);

  keys_kernel<<<dim3(HH / 256, LQ / 16, BB), 128, 0, stream>>>(q, Wt, bias, keys);

  attn_kernel<<<dim3(LP / 16, BB), 128, 0, stream>>>(p, qT, keys, out);
}
